// AdaptiveComputationTime_30889404793001
// MI455X (gfx1250) — matrix operand order repaired and checked by reading, not compiled, not run on hardware
//
#include <hip/hip_runtime.h>
#include <hip/hip_bf16.h>

// ACT halting step, fused single-pass kernel for gfx1250 (MI455X).
//
// Roofline: ~1.07 GB compulsory HBM traffic / 23.3 TB/s => ~46 us floor;
// compute (~0.5 GFLOP) is irrelevant. Strategy: one block per 16 tokens,
// WMMA-based batched dot (h@W) on the first read of h, then elementwise
// phase re-reads the 64KB h tile from L2 (192MB) so h costs HBM once.

typedef __attribute__((ext_vector_type(16))) _Float16 v16h;
typedef __attribute__((ext_vector_type(8)))  float    v8f;
typedef __attribute__((ext_vector_type(4)))  float    v4f;   // native vector for NT builtins
typedef __attribute__((ext_vector_type(2)))  float    v2f;

#define HDIM 1024     // hidden size H
#define TPB  16       // tokens per block (one WMMA tile of rows)
#define THREADS 256   // 8 waves (wave32)

__global__ __launch_bounds__(THREADS)
void act_fused_kernel(const float* __restrict__ h,
                      const float* __restrict__ wh,
                      const float* __restrict__ acc_p,
                      const float* __restrict__ rem,
                      const float* __restrict__ exi,
                      const int*   __restrict__ run,
                      const float* __restrict__ W,
                      const float* __restrict__ bvec,
                      const float* __restrict__ coeffv,
                      const int*   __restrict__ stepv,
                      float* __restrict__ out_h,
                      float* __restrict__ out_wh,
                      float* __restrict__ out_accp,
                      float* __restrict__ out_rem,
                      float* __restrict__ out_exit,
                      float* __restrict__ out_run)
{
    __shared__ float s_part[8][TPB];   // per-wave partial dots
    __shared__ float s_scale[TPB];     // run_f * mask_continue   (h_out scale)
    __shared__ float s_updv[TPB];      // run_f * update          (weighted_h FMA scale)

    const int tid  = threadIdx.x;
    const int wave = tid >> 5;
    const int lane = tid & 31;
    const long long tok0 = (long long)blockIdx.x * TPB;

    // ================= Pass 1: 16 dot products via v_wmma_f32_16x16x32_f16 ====
    // Wave w owns K in [w*128, w*128+128): 4 chunks of K=32.
    // A layout (16x32 f16, ISA 7.12.2): lane m=L&15 is row M; hi=L>>4;
    //   VGPR v holds K pair at (v>>2)*16 + hi*8 + (v&3)*2.
    // B layout (32x16 f16, dense-B per ISA 7.12.4): lane n=L&15 is col N; hi=L>>4;
//   VGPR v holds K pair at (v>>2)*16 + hi*8 + (v&3)*2 (same as A).  W goes into column N=0 only.
    const int m  = lane & 15;
    const int hi = lane >> 4;
    const float* hrow = h + (tok0 + m) * HDIM;

    v8f c = {};
    const int kWave0 = wave * 128;
    #pragma unroll
    for (int chunk = 0; chunk < 4; ++chunk) {
        const int kc = kWave0 + chunk * 32;
        v16h a;
        #pragma unroll
        for (int v = 0; v < 8; ++v) {
            const int kp = kc + ((v >> 2) << 4) + (hi << 3) + ((v & 3) << 1);
            const v2f hv = *(const v2f*)(hrow + kp);
            a[2 * v]     = (_Float16)hv.x;
            a[2 * v + 1] = (_Float16)hv.y;
        }
        v16h bm = {};
        if (m == 0) {                       // column 0 of B carries W
            #pragma unroll
            for (int v = 0; v < 8; ++v) {
                const int kp = kc + ((v >> 2) << 4) + (hi << 3) + ((v & 3) << 1);
                const v2f wv = *(const v2f*)(W + kp);
                bm[2 * v]     = (_Float16)wv.x;
                bm[2 * v + 1] = (_Float16)wv.y;
            }
        }
        // D = A*B + C, f32 accumulate
        c = __builtin_amdgcn_wmma_f32_16x16x32_f16(
                /*neg_a=*/false, a, /*neg_b=*/false, bm,
                /*c_mod=*/(short)0, c, /*reuse_a=*/false, /*reuse_b=*/false);
    }
    // Column N=0 of C: lane 0 holds rows 0..7 in c[0..7], lane 16 holds rows 8..15.
    if (m == 0) {
        #pragma unroll
        for (int j = 0; j < 8; ++j)
            s_part[wave][(hi << 3) + j] = c[j];
    }
    __syncthreads();

    // ================= Per-token scalar ACT math (threads 0..15) =============
    if (tid < TPB) {
        float dot = 0.f;
        #pragma unroll
        for (int wv = 0; wv < 8; ++wv) dot += s_part[wv][tid];

        const long long t = tok0 + tid;
        const int   runi  = run[t];
        const float run_f = (runi > 0) ? 1.f : 0.f;
        const float ap    = acc_p[t];
        const float logit = dot * run_f + bvec[0];       // (h*run)@W + b
        const float p     = (1.f / (1.f + __expf(-logit))) * coeffv[0] * run_f;

        const bool  run_b = runi > 0;
        const bool  mc    = ((ap + p) < 0.99f) && run_b; // continue mask
        const bool  me    = (!mc) && run_b;              // exit mask
        const float mcf   = mc ? 1.f : 0.f;
        const float mef   = me ? 1.f : 0.f;

        const float update  = p * mcf + (1.f - ap) * mef;
        const float ap_new  = ap + update * mcf;
        const float rem_new = rem[t] + (1.f - ap_new) * mef;
        const float ex_new  = exi[t] + (float)stepv[0] * mef;
        const int   run_new = runi * (mc ? 1 : 0);

        out_accp[t] = ap_new;
        out_rem[t]  = rem_new;
        out_exit[t] = ex_new;
        out_run[t]  = (float)run_new;    // int32 result stored numerically

        s_scale[tid] = run_f * mcf;      // h_out   = h * (run_f*mc)
        s_updv[tid]  = run_f * update;   // wh_out  = h * (run_f*update) + wh
    }
    __syncthreads();

    // ================= Pass 2: elementwise stream, 128-bit, NT hints =========
    // h tile (64KB) is L2-hot from pass 1; weighted_h streamed NT (read-once);
    // both outputs stored NT (write-once, never re-read).
    const long long base = tok0 * HDIM;
    const v4f* h4 = (const v4f*)(h  + base);
    const v4f* w4 = (const v4f*)(wh + base);
    v4f* oh4 = (v4f*)(out_h  + base);
    v4f* ow4 = (v4f*)(out_wh + base);

    #pragma unroll 4
    for (int i = 0; i < (TPB * HDIM / 4) / THREADS; ++i) {   // 16 iterations
        const int e = tid + i * THREADS;       // float4 index in tile, 0..4095
        const int t = e >> 8;                  // HDIM/4 = 256 float4 per token
        const float so = s_scale[t];
        const float su = s_updv[t];

        const v4f hv = h4[e];                                  // L2 hit
        const v4f wv = __builtin_nontemporal_load(&w4[e]);     // stream in

        v4f ho, wo;
        ho.x = hv.x * so;               ho.y = hv.y * so;
        ho.z = hv.z * so;               ho.w = hv.w * so;
        wo.x = fmaf(hv.x, su, wv.x);    wo.y = fmaf(hv.y, su, wv.y);
        wo.z = fmaf(hv.z, su, wv.z);    wo.w = fmaf(hv.w, su, wv.w);

        __builtin_nontemporal_store(ho, &oh4[e]);              // stream out
        __builtin_nontemporal_store(wo, &ow4[e]);
    }
}

extern "C" void kernel_launch(void* const* d_in, const int* in_sizes, int n_in,
                              void* d_out, int out_size, void* d_ws, size_t ws_size,
                              hipStream_t stream) {
    // setup_inputs() order:
    // 0:h 1:weighted_h 2:acc_p 3:remainders 4:exit_ 5:run 6:W 7:b 8:coeff 9:step
    const float* h   = (const float*)d_in[0];
    const float* wh  = (const float*)d_in[1];
    const float* ap  = (const float*)d_in[2];
    const float* rem = (const float*)d_in[3];
    const float* ex  = (const float*)d_in[4];
    const int*   run = (const int*)  d_in[5];
    const float* W   = (const float*)d_in[6];
    const float* b   = (const float*)d_in[7];
    const float* cf  = (const float*)d_in[8];
    const int*   st  = (const int*)  d_in[9];

    const long long NT  = in_sizes[2];                 // B*M tokens (65536)
    const long long BIG = (long long)in_sizes[0];      // B*M*H

    float* out      = (float*)d_out;
    float* out_h    = out;
    float* out_wh   = out + BIG;
    float* out_ap   = out + 2 * BIG;
    float* out_rem  = out_ap  + NT;
    float* out_exit = out_rem + NT;
    float* out_run  = out_exit + NT;

    dim3 grid((unsigned)(NT / TPB));   // 4096 blocks
    dim3 block(THREADS);               // 256 threads = 8 wave32
    act_fused_kernel<<<grid, block, 0, stream>>>(
        h, wh, ap, rem, ex, run, W, b, cf, st,
        out_h, out_wh, out_ap, out_rem, out_exit, out_run);
}
